// Layer_47691316855120
// MI455X (gfx1250) — compile-verified, matrix operand order checked by reading, not run on hardware
//
#include <hip/hip_runtime.h>
#include <hip/hip_bf16.h>
#include <math.h>
#include <cstdint>

typedef __bf16 bf16_t;
typedef bf16_t v16bf __attribute__((ext_vector_type(16)));
typedef float  v8f   __attribute__((ext_vector_type(8)));
typedef int    v4i   __attribute__((ext_vector_type(4)));

#define AS1 __attribute__((address_space(1)))
#define AS3 __attribute__((address_space(3)))

#if __has_builtin(__builtin_amdgcn_global_load_async_to_lds_b128) && \
    __has_builtin(__builtin_amdgcn_s_wait_asynccnt)
#define USE_ASYNC 1
#else
#define USE_ASYNC 0
#endif

__device__ __forceinline__ void async_cp16(const void* g, void* l) {
#if USE_ASYNC
  // GLOBAL_LOAD_ASYNC_TO_LDS_B128: per-lane global gather -> LDS, ASYNCcnt.
  __builtin_amdgcn_global_load_async_to_lds_b128(
      (AS1 v4i*)(uintptr_t)g,
      (AS3 v4i*)(unsigned int)(uintptr_t)l, 0, 0);
#else
  *(uint4*)l = *(const uint4*)g;
#endif
}

__device__ __forceinline__ void async_wait() {
#if USE_ASYNC
  __builtin_amdgcn_s_wait_asynccnt(0);
#endif
}

__device__ __forceinline__ v8f zero8() {
  v8f z = {0.f,0.f,0.f,0.f,0.f,0.f,0.f,0.f};
  return z;
}

// ---- DPP16 cross-lane reductions over each 16-lane half (wave32) ----------
// quad_perm xor1 (0xB1), quad_perm xor2 (0x4E), row_ror:4 (0x124), row_ror:8
// (0x128) -> full 16-lane reduce in 4 VALU ops, no LDS, no waits.
template <int CTRL>
__device__ __forceinline__ float dpp_mov(float x) {
  return __builtin_bit_cast(
      float, __builtin_amdgcn_update_dpp(
                 0, __builtin_bit_cast(int, x), CTRL, 0xF, 0xF, true));
}
__device__ __forceinline__ float row_max16(float x) {
  x = fmaxf(x, dpp_mov<0xB1>(x));
  x = fmaxf(x, dpp_mov<0x4E>(x));
  x = fmaxf(x, dpp_mov<0x124>(x));
  x = fmaxf(x, dpp_mov<0x128>(x));
  return x;
}
__device__ __forceinline__ float row_sum16(float x) {
  x += dpp_mov<0xB1>(x);
  x += dpp_mov<0x4E>(x);
  x += dpp_mov<0x124>(x);
  x += dpp_mov<0x128>(x);
  return x;
}

// Load a 16x32 bf16 fragment (CDNA5 16-bit matrix VGPR layout: lane 0-15 row
// M=lane with K=0..7,16..23; lane 16-31 same row, K=8..15,24..31) from a
// row-major [rows][stride] bf16 buffer.
template <typename T>
__device__ __forceinline__ v16bf load_frag(const T* base, int stride_elems, int lane) {
  const int r = lane & 15;
  const int h = lane >> 4;
  union { v16bf v; uint4 q[2]; } f;
  const T* p = base + r * stride_elems + h * 8;
  f.q[0] = *(const uint4*)(p);
  f.q[1] = *(const uint4*)(p + 16);
  return f.v;
}

// ---------------------------------------------------------------------------
// Small prologue kernels
// ---------------------------------------------------------------------------
__global__ void silu_kernel(const float* __restrict__ c, float* __restrict__ sc, int n) {
  int i = blockIdx.x * 256 + threadIdx.x;
  if (i < n) { float x = c[i]; sc[i] = x / (1.0f + expf(-x)); }
}

__global__ void modmm_kernel(const float* __restrict__ sc, const float* __restrict__ w,
                             const float* __restrict__ bmod, float* __restrict__ mod) {
  const int j = blockIdx.x * 256 + threadIdx.x;  // 0..6143
  const int b = blockIdx.y;                      // 0..7
  float acc = bmod[j];
  const float* srow = sc + b * 1024;
  for (int k = 0; k < 1024; ++k)
    acc += srow[k] * w[(size_t)k * 6144 + j];
  mod[b * 6144 + j] = acc;
}

// fp32 w[K][N] -> bf16 wT[N][K]
__global__ void convT_kernel(const float* __restrict__ w, bf16_t* __restrict__ wT,
                             int K, int N) {
  size_t idx = (size_t)blockIdx.x * 256 + threadIdx.x;
  int k = (int)(idx / N);
  int n = (int)(idx % N);
  wT[(size_t)n * K + k] = (bf16_t)w[idx];
}

// LayerNorm + adaLN modulate -> bf16.  One 256-thread block per token row.
__global__ void ln_modulate_kernel(const float* __restrict__ xin,
                                   const float* __restrict__ mod,
                                   int shift_off, int scale_off,
                                   bf16_t* __restrict__ out) {
  __shared__ float redS[16];
  __shared__ float redQ[16];
  const int row = blockIdx.x;   // 0..8191
  const int b   = row >> 10;
  const int tid = threadIdx.x;

  float vals[4];
  float s = 0.f, ss = 0.f;
#pragma unroll
  for (int i = 0; i < 4; ++i) {
    float x = xin[(size_t)row * 1024 + tid + i * 256];
    vals[i] = x; s += x; ss += x * x;
  }
  s  = row_sum16(s);
  ss = row_sum16(ss);
  const int g = tid >> 4;             // 16 groups of 16 lanes
  if ((tid & 15) == 0) { redS[g] = s; redQ[g] = ss; }
  __syncthreads();
  s = 0.f; ss = 0.f;
#pragma unroll
  for (int w = 0; w < 16; ++w) { s += redS[w]; ss += redQ[w]; }

  const float mu   = s * (1.0f / 1024.0f);
  const float var  = ss * (1.0f / 1024.0f) - mu * mu;
  const float rstd = rsqrtf(var + 1e-6f);
  const float* mrow = mod + b * 6144;
#pragma unroll
  for (int i = 0; i < 4; ++i) {
    int c = tid + i * 256;
    float y = ((vals[i] - mu) * rstd) * (1.0f + mrow[scale_off + c]) + mrow[shift_off + c];
    out[(size_t)row * 1024 + c] = (bf16_t)y;
  }
}

// ---------------------------------------------------------------------------
// bf16 NT GEMM: C[M][N] = A[M][K] * Bt[N][K]^T, fp32 accum via WMMA.
// 256 threads (8 waves), 128x128 tile, wave = 64x32 (4x2 WMMA subtiles).
// Double-buffered LDS staging via GLOBAL_LOAD_ASYNC_TO_LDS_B128 (ASYNCcnt).
// EPI: 0=qkv split, 1=proj+gate+residual, 2=fc1 gelu, 3=fc2+gate+residual
// ---------------------------------------------------------------------------
template <int EPI>
__global__ __launch_bounds__(256, 2) void gemm_bf16_kernel(
    const bf16_t* __restrict__ A, const bf16_t* __restrict__ Bt,
    const int M, const int N, const int K,
    const float* __restrict__ bias,
    const float* __restrict__ resid,
    const float* __restrict__ modrow,
    float* __restrict__ outf,
    bf16_t* __restrict__ outb,
    bf16_t* __restrict__ outq, bf16_t* __restrict__ outk, bf16_t* __restrict__ outv) {
  __shared__ __align__(16) bf16_t As[2][128 * 40];
  __shared__ __align__(16) bf16_t Bs[2][128 * 40];

  const int tid   = threadIdx.x;
  const int lane  = tid & 31;
  const int wave  = tid >> 5;
  const int waveM = wave >> 2;   // 0..1
  const int waveN = wave & 3;    // 0..3
  const int m0 = blockIdx.y * 128;
  const int n0 = blockIdx.x * 128;

  // this thread's two 16-byte staging chunks (row, col within 128x32 tile)
  const int r0 = tid >> 2, c0 = (tid & 3) << 3;
  const int r1 = (tid + 256) >> 2, c1 = ((tid + 256) & 3) << 3;

  auto issue_tile = [&](int buf, int k0) {
    async_cp16(&A[(size_t)(m0 + r0) * K + k0 + c0], &As[buf][r0 * 40 + c0]);
    async_cp16(&A[(size_t)(m0 + r1) * K + k0 + c1], &As[buf][r1 * 40 + c1]);
    async_cp16(&Bt[(size_t)(n0 + r0) * K + k0 + c0], &Bs[buf][r0 * 40 + c0]);
    async_cp16(&Bt[(size_t)(n0 + r1) * K + k0 + c1], &Bs[buf][r1 * 40 + c1]);
  };

  v8f acc[4][2];
#pragma unroll
  for (int mt = 0; mt < 4; ++mt)
#pragma unroll
    for (int nt = 0; nt < 2; ++nt) acc[mt][nt] = zero8();

  const int ksteps = K >> 5;
  issue_tile(0, 0);
  for (int ki = 0; ki < ksteps; ++ki) {
    const int cur = ki & 1;
    async_wait();        // my async loads (for `cur`) have landed in LDS
    __syncthreads();     // everyone's loads landed; prior reads of `cur^1` done
    if (ki + 1 < ksteps) issue_tile(cur ^ 1, (ki + 1) << 5);

    v16bf aF[4], bF[2];
#pragma unroll
    for (int mt = 0; mt < 4; ++mt)
      aF[mt] = load_frag(&As[cur][(waveM * 64 + mt * 16) * 40], 40, lane);
#pragma unroll
    for (int nt = 0; nt < 2; ++nt)
      bF[nt] = load_frag(&Bs[cur][(waveN * 32 + nt * 16) * 40], 40, lane);
#pragma unroll
    for (int mt = 0; mt < 4; ++mt)
#pragma unroll
      for (int nt = 0; nt < 2; ++nt)
        acc[mt][nt] = __builtin_amdgcn_wmma_f32_16x16x32_bf16(
            false, aF[mt], false, bF[nt], (short)0, acc[mt][nt], false, false);
    __syncthreads();     // reads of `cur` done before it is refilled
  }

  const int half = lane >> 4;
  const int cN   = lane & 15;
#pragma unroll
  for (int mt = 0; mt < 4; ++mt) {
#pragma unroll
    for (int nt = 0; nt < 2; ++nt) {
#pragma unroll
      for (int j = 0; j < 8; ++j) {
        const int m = m0 + waveM * 64 + mt * 16 + j + 8 * half;
        const int n = n0 + waveN * 32 + nt * 16 + cN;
        float val = acc[mt][nt][j];
        if constexpr (EPI == 0) {            // split qkv -> [b][h][n][d] bf16
          const int which = n >> 10;
          const int hh = (n & 1023) >> 6;
          const int d  = n & 63;
          const int bb = m >> 10;
          const int nn = m & 1023;
          const size_t dst = ((size_t)(bb * 16 + hh) * 1024 + nn) * 64 + d;
          if (which == 0)      outq[dst] = (bf16_t)(val * 0.125f);  // 1/sqrt(64)
          else if (which == 1) outk[dst] = (bf16_t)val;
          else                 outv[dst] = (bf16_t)val;
        } else if constexpr (EPI == 1) {     // x_mid = x + g_msa*(o@Wp + b)
          const int bb = m >> 10;
          const float g = modrow[bb * 6144 + 2048 + n];
          outf[(size_t)m * 1024 + n] =
              resid[(size_t)m * 1024 + n] + g * (val + bias[n]);
        } else if constexpr (EPI == 2) {     // h = gelu(x2@W1 + b) (tanh)
          float u = val + bias[n];
          float t = tanhf(0.7978845608028654f * (u + 0.044715f * u * u * u));
          outb[(size_t)m * 4096 + n] = (bf16_t)(0.5f * u * (1.0f + t));
        } else {                             // out = x_mid + g_mlp*(h@W2 + b)
          const int bb = m >> 10;
          const float g = modrow[bb * 6144 + 5120 + n];
          outf[(size_t)m * 1024 + n] =
              resid[(size_t)m * 1024 + n] + g * (val + bias[n]);
        }
      }
    }
  }
}

// ---------------------------------------------------------------------------
// Flash attention. Grid (8 qblocks, 128 batch*head), block 256 (8 waves).
// Each wave owns 16 query rows; 64-key blocks; online softmax with DPP16
// row reductions matching the WMMA C-layout (row = vgpr_idx + 8*half).
// ---------------------------------------------------------------------------
__global__ __launch_bounds__(256, 2) void attention_kernel(
    const bf16_t* __restrict__ q, const bf16_t* __restrict__ k,
    const bf16_t* __restrict__ v, bf16_t* __restrict__ o) {
  __shared__ __align__(16) bf16_t Ks [64 * 72];       // [key][d]
  __shared__ __align__(16) bf16_t Vts[64 * 72];       // [d][key]
  __shared__ __align__(16) bf16_t Ps [8 * 16 * 72];   // per-wave P tile

  const int tid  = threadIdx.x;
  const int lane = tid & 31;
  const int wave = tid >> 5;
  const int bh   = blockIdx.y;
  const int bb   = bh >> 4;
  const int hh   = bh & 15;
  const int qrow0 = blockIdx.x * 128 + wave * 16;

  const bf16_t* qp = q + (size_t)bh * 1024 * 64;
  const bf16_t* kp = k + (size_t)bh * 1024 * 64;
  const bf16_t* vp = v + (size_t)bh * 1024 * 64;

  v16bf qa[2];
  qa[0] = load_frag(qp + (size_t)qrow0 * 64,      64, lane);
  qa[1] = load_frag(qp + (size_t)qrow0 * 64 + 32, 64, lane);

  v8f oacc[4];
#pragma unroll
  for (int dt = 0; dt < 4; ++dt) oacc[dt] = zero8();
  float m_run[8], l_run[8];
#pragma unroll
  for (int j = 0; j < 8; ++j) { m_run[j] = -1e30f; l_run[j] = 0.0f; }

  bf16_t* Pw = &Ps[wave * 16 * 72];
  const int half = lane >> 4;
  const int cN   = lane & 15;
  const float LOG2E = 1.44269504088896f;

  // K-tile staging chunks for this thread (64 rows x 64 cols, 8 chunks/row)
  const int kr0 = tid >> 3, kc0 = (tid & 7) << 3;
  const int kr1 = (tid + 256) >> 3, kc1 = ((tid + 256) & 7) << 3;
  // V transpose: this thread copies 16 consecutive d's of one key row
  const int vkey = tid >> 2, vd0 = (tid & 3) << 4;

  for (int kb = 0; kb < 16; ++kb) {
    __syncthreads();
    async_cp16(&kp[(size_t)(kb * 64 + kr0) * 64 + kc0], &Ks[kr0 * 72 + kc0]);
    async_cp16(&kp[(size_t)(kb * 64 + kr1) * 64 + kc1], &Ks[kr1 * 72 + kc1]);
    {
      const bf16_t* src = &vp[(size_t)(kb * 64 + vkey) * 64 + vd0];
#pragma unroll
      for (int i = 0; i < 16; ++i)
        Vts[(vd0 + i) * 72 + vkey] = src[i];
    }
    async_wait();
    __syncthreads();

    // S = Q K^T  (16 x 64 per wave)
    v8f s[4];
#pragma unroll
    for (int nt = 0; nt < 4; ++nt) s[nt] = zero8();
#pragma unroll
    for (int ks = 0; ks < 2; ++ks)
#pragma unroll
      for (int nt = 0; nt < 4; ++nt) {
        v16bf bF = load_frag(&Ks[nt * 16 * 72 + ks * 32], 72, lane);
        s[nt] = __builtin_amdgcn_wmma_f32_16x16x32_bf16(
            false, qa[ks], false, bF, (short)0, s[nt], false, false);
      }

    // online softmax (rows j + 8*half; DPP reduce across this half's 16 lanes)
    float scale_j[8];
#pragma unroll
    for (int j = 0; j < 8; ++j) {
      float mx = fmaxf(fmaxf(s[0][j], s[1][j]), fmaxf(s[2][j], s[3][j]));
      mx = row_max16(mx);
      float mnew = fmaxf(m_run[j], mx);
      scale_j[j] = exp2f((m_run[j] - mnew) * LOG2E);
      m_run[j] = mnew;
      float rs = 0.0f;
#pragma unroll
      for (int nt = 0; nt < 4; ++nt) {
        float p = exp2f((s[nt][j] - mnew) * LOG2E);
        s[nt][j] = p;
        rs += p;
      }
      rs = row_sum16(rs);
      l_run[j] = l_run[j] * scale_j[j] + rs;
    }
#pragma unroll
    for (int dt = 0; dt < 4; ++dt)
#pragma unroll
      for (int j = 0; j < 8; ++j) oacc[dt][j] *= scale_j[j];

    // P (C-layout) -> LDS row-major, so it can be reloaded as an A fragment
#pragma unroll
    for (int nt = 0; nt < 4; ++nt)
#pragma unroll
      for (int j = 0; j < 8; ++j)
        Pw[(j + 8 * half) * 72 + nt * 16 + cN] = (bf16_t)s[nt][j];
    asm volatile("" ::: "memory");   // keep LDS writes before reads

    // O += P @ V
#pragma unroll
    for (int ks = 0; ks < 2; ++ks) {
      v16bf pa = load_frag(&Pw[ks * 32], 72, lane);
#pragma unroll
      for (int dt = 0; dt < 4; ++dt) {
        v16bf bv = load_frag(&Vts[dt * 16 * 72 + ks * 32], 72, lane);
        oacc[dt] = __builtin_amdgcn_wmma_f32_16x16x32_bf16(
            false, pa, false, bv, (short)0, oacc[dt], false, false);
      }
    }
  }

  // O /= l, write row-major [b][n][h*64+d] so proj GEMM consumes it directly
#pragma unroll
  for (int j = 0; j < 8; ++j) {
    float inv = 1.0f / l_run[j];
    int n = qrow0 + j + 8 * half;
#pragma unroll
    for (int dt = 0; dt < 4; ++dt) {
      int d = dt * 16 + cN;
      o[((size_t)(bb * 1024 + n)) * 1024 + hh * 64 + d] = (bf16_t)(oacc[dt][j] * inv);
    }
  }
}

// ---------------------------------------------------------------------------
extern "C" void kernel_launch(void* const* d_in, const int* in_sizes, int n_in,
                              void* d_out, int out_size, void* d_ws, size_t ws_size,
                              hipStream_t stream) {
  const float* x      = (const float*)d_in[0];
  const float* c      = (const float*)d_in[1];
  const float* w_mod  = (const float*)d_in[2];
  const float* b_mod  = (const float*)d_in[3];
  const float* w_qkv  = (const float*)d_in[4];
  const float* w_proj = (const float*)d_in[5];
  const float* b_proj = (const float*)d_in[6];
  const float* w_fc1  = (const float*)d_in[7];
  const float* b_fc1  = (const float*)d_in[8];
  const float* w_fc2  = (const float*)d_in[9];
  const float* b_fc2  = (const float*)d_in[10];
  float* out = (float*)d_out;

  char* p = (char*)d_ws;
  auto take = [&p](size_t bytes) {
    void* r = (void*)p;
    p += (bytes + 255) & ~(size_t)255;
    return r;
  };
  bf16_t* wqkvT = (bf16_t*)take((size_t)3072 * 1024 * 2);
  bf16_t* wprojT= (bf16_t*)take((size_t)1024 * 1024 * 2);
  bf16_t* wfc1T = (bf16_t*)take((size_t)4096 * 1024 * 2);
  bf16_t* wfc2T = (bf16_t*)take((size_t)1024 * 4096 * 2);
  float*  siluc = (float*) take((size_t)8 * 1024 * 4);
  float*  modb  = (float*) take((size_t)8 * 6144 * 4);
  bf16_t* x1    = (bf16_t*)take((size_t)8192 * 1024 * 2);
  bf16_t* qb    = (bf16_t*)take((size_t)8192 * 1024 * 2);
  bf16_t* kb    = (bf16_t*)take((size_t)8192 * 1024 * 2);
  bf16_t* vb    = (bf16_t*)take((size_t)8192 * 1024 * 2);
  bf16_t* ob    = (bf16_t*)take((size_t)8192 * 1024 * 2);
  float*  xmid  = (float*) take((size_t)8192 * 1024 * 4);
  bf16_t* x2    = (bf16_t*)take((size_t)8192 * 1024 * 2);
  bf16_t* hb    = (bf16_t*)take((size_t)8192 * 4096 * 2);

  // 1) modulation vector: silu(c) @ w_mod + b_mod
  silu_kernel<<<32, 256, 0, stream>>>(c, siluc, 8192);
  modmm_kernel<<<dim3(24, 8), 256, 0, stream>>>(siluc, w_mod, b_mod, modb);

  // 2) weights -> bf16 transposed [N][K]
  convT_kernel<<<(1024 * 3072) / 256, 256, 0, stream>>>(w_qkv,  wqkvT, 1024, 3072);
  convT_kernel<<<(1024 * 1024) / 256, 256, 0, stream>>>(w_proj, wprojT, 1024, 1024);
  convT_kernel<<<(1024 * 4096) / 256, 256, 0, stream>>>(w_fc1,  wfc1T, 1024, 4096);
  convT_kernel<<<(4096 * 1024) / 256, 256, 0, stream>>>(w_fc2,  wfc2T, 4096, 1024);

  // 3) x1 = modulate(LN(x), sh_msa, sc_msa)
  ln_modulate_kernel<<<8192, 256, 0, stream>>>(x, modb, 0, 1024, x1);

  // 4) qkv = x1 @ w_qkv, split to q/k/v planes (q pre-scaled)
  gemm_bf16_kernel<0><<<dim3(24, 64), 256, 0, stream>>>(
      x1, wqkvT, 8192, 3072, 1024, nullptr, nullptr, nullptr,
      nullptr, nullptr, qb, kb, vb);

  // 5) attention
  attention_kernel<<<dim3(8, 128), 256, 0, stream>>>(qb, kb, vb, ob);

  // 6) x_mid = x + g_msa * (o @ w_proj + b_proj)
  gemm_bf16_kernel<1><<<dim3(8, 64), 256, 0, stream>>>(
      ob, wprojT, 8192, 1024, 1024, b_proj, x, modb,
      xmid, nullptr, nullptr, nullptr, nullptr);

  // 7) x2 = modulate(LN(x_mid), sh_mlp, sc_mlp)
  ln_modulate_kernel<<<8192, 256, 0, stream>>>(xmid, modb, 3072, 4096, x2);

  // 8) h = gelu(x2 @ w_fc1 + b_fc1)
  gemm_bf16_kernel<2><<<dim3(32, 64), 256, 0, stream>>>(
      x2, wfc1T, 8192, 4096, 1024, b_fc1, nullptr, nullptr,
      nullptr, hb, nullptr, nullptr, nullptr);

  // 9) out = x_mid + g_mlp * (h @ w_fc2 + b_fc2)
  gemm_bf16_kernel<3><<<dim3(8, 64), 256, 0, stream>>>(
      hb, wfc2T, 8192, 1024, 4096, b_fc2, xmid, modb,
      out, nullptr, nullptr, nullptr, nullptr);
}
